// WeatherLSTM_31602369364549
// MI455X (gfx1250) — compile-verified, matrix operand order checked by reading, not run on hardware
//
#include <hip/hip_runtime.h>
#include <hip/hip_bf16.h>

// ---------------------------------------------------------------------------
// WeatherLSTM on MI455X (gfx1250).
//  * Layer1 x_t == h  =>  gates = h @ (Wih1+Whh1)^T  (pre-summed, K=1024).
//  * Weights packed once to bf16 in WMMA B-fragment order (~17MB, L2-resident).
//  * Activations kept in bf16; GEMM A-tiles staged with one async global->LDS
//    b128 copy per thread per 64-K group (ASYNCcnt), double-buffered LDS,
//    next group's copy issued before computing the current one.
//  * Register blocking: each wave computes 2 M-tiles x 2 N-tiles -> every A and
//    every B fragment feeds 2 WMMAs (8 v_wmma : 8 global b128 : 8 ds b128 per
//    group), halving L2 weight traffic vs 1x2 blocking (GEMM is L2-bound).
//  * Cell + gradient-threshold fused (LDS neighbor exchange, block-edge
//    neighbor recomputed from gates); c ping-pongs between buffers.
// ---------------------------------------------------------------------------

typedef __attribute__((ext_vector_type(16))) __bf16 v16bf;
typedef __attribute__((ext_vector_type(8)))  __bf16 v8bf;
typedef __attribute__((ext_vector_type(8)))  float  v8f;

#define HDIM   1024
#define NGATE  4096      // 4*H
#define BATCH  256
#define TSTEPS 128
#define IDIM   64
#define KGROUP 64        // K staged per barrier round = 2 WMMA chunks

__device__ __forceinline__ unsigned short f2bf(float f) {
    unsigned int u = __builtin_bit_cast(unsigned int, f);
    unsigned int r = u + 0x7FFFu + ((u >> 16) & 1u);   // round-to-nearest-even
    return (unsigned short)(r >> 16);
}

// --------------------------------------------------------------------------
// Weight packing: B-matrix 32(K)x16(N) bf16 fragment order (ISA 7.12.2).
// Flat element r in [0,512): lane = r/16, e = r%16.
//   n = ntile*16 + lane%16 ; base = (lane/16)*8
//   k = ktile*32 + (e<8 ? base+e : 16+base+(e-8))
// -> each lane's 16 bf16 contiguous: 2x global_load_b128 in the GEMM.
// --------------------------------------------------------------------------
__device__ __forceinline__ void frag_coords(long gid, int& k, int& n) {
    int r      = (int)(gid & 511);
    long tile  = gid >> 9;
    int ntile  = (int)(tile & 255);
    int ktile  = (int)(tile >> 8);
    int lane   = r >> 4, e = r & 15;
    n = ntile * 16 + (lane & 15);
    int base = (lane >> 4) * 8;
    int kk = (e < 8) ? (base + e) : (16 + base + (e - 8));
    k = ktile * 32 + kk;
}

__global__ void pack_w_concat(const float* __restrict__ W0, int K0,
                              const float* __restrict__ W1, int K1,
                              unsigned short* __restrict__ dst, long total) {
    long gid = (long)blockIdx.x * blockDim.x + threadIdx.x;
    if (gid >= total) return;
    int k, n; frag_coords(gid, k, n);
    float v = (k < K0) ? W0[(long)n * K0 + k] : W1[(long)n * K1 + (k - K0)];
    dst[gid] = f2bf(v);
}

__global__ void pack_w_sum(const float* __restrict__ A, const float* __restrict__ B,
                           int K, unsigned short* __restrict__ dst, long total) {
    long gid = (long)blockIdx.x * blockDim.x + threadIdx.x;
    if (gid >= total) return;
    int k, n; frag_coords(gid, k, n);
    dst[gid] = f2bf(A[(long)n * K + k] + B[(long)n * K + k]);
}

__global__ void pack_f32_to_bf16(const float* __restrict__ src,
                                 unsigned short* __restrict__ dst, int n) {
    int i = blockIdx.x * blockDim.x + threadIdx.x;
    if (i < n) dst[i] = f2bf(src[i]);
}

__global__ void add_bias_vec(const float* __restrict__ a, const float* __restrict__ b,
                             float* __restrict__ o, int n) {
    int i = blockIdx.x * blockDim.x + threadIdx.x;
    if (i < n) o[i] = a[i] + b[i];
}

__global__ void zero_u32(unsigned int* __restrict__ p, int n) {
    int i = blockIdx.x * blockDim.x + threadIdx.x;
    if (i < n) p[i] = 0u;
}

// --------------------------------------------------------------------------
// GEMM: gates[BATCH x 4096] = [A0 | A1](bf16) @ Wpack^T + bias
// Block = 256 threads = 8 waves. blockIdx.y owns a 32-row M block (2 tiles);
// wave owns 2 N-tiles -> wave computes a 32x32 output block (4 accumulators).
// 64-K groups staged via one async b128 per thread, double-buffered.
// --------------------------------------------------------------------------
__global__ __launch_bounds__(256)
void gemm_gates(const unsigned short* __restrict__ A0, int lda0, int K0,
                const unsigned short* __restrict__ A1, int lda1, int K1,
                const unsigned short* __restrict__ Bpack,
                const float* __restrict__ bias,
                float* __restrict__ gates) {
    __shared__ __align__(16) unsigned short aT[2][32 * KGROUP];   // 2 x 4KB

    const int tid    = threadIdx.x;
    const int lane   = tid & 31;
    const int wave   = tid >> 5;                     // 0..7
    const int mbase  = blockIdx.y * 32;              // 2 M-tiles
    const int ntile0 = (blockIdx.x * 8 + wave) * 2;  // 2 N-tiles
    const int nGroups = (K0 + K1) / KGROUP;          // 17 (L0) or 16 (L1)

    // staging: 32 rows x 64 K = 4KB; one async b128 (8 bf16) per thread
    const int srow = tid >> 3;                       // 0..31
    const int sk   = (tid & 7) * 8;                  // 0..56
    // LDS byte offsets (low 32 bits of generic pointer == LDS offset)
    const unsigned ldsDst0 =
        (unsigned)(unsigned long long)&aT[0][srow * KGROUP + sk];
    const unsigned ldsDst1 =
        (unsigned)(unsigned long long)&aT[1][srow * KGROUP + sk];

    auto stage = [&](int g) {
        int kg = g * KGROUP + sk;
        const unsigned short* src =
            (kg < K0) ? A0 + (long)(mbase + srow) * lda0 + kg
                      : A1 + (long)(mbase + srow) * lda1 + (kg - K0);
        unsigned dst = (g & 1) ? ldsDst1 : ldsDst0;
        asm volatile("global_load_async_to_lds_b128 %0, %1, off"
                     :: "v"(dst), "v"(src) : "memory");
    };

    // A-fragment coords (16-bit A 16x32 layout, ISA 7.12.2)
    const int m = lane & 15, half = lane >> 4;

    stage(0);
    v8f acc00 = {}, acc01 = {}, acc10 = {}, acc11 = {};
    for (int g = 0; g < nGroups; ++g) {
        asm volatile("s_wait_asynccnt 0x0" ::: "memory");
        __syncthreads();                 // group g staged; prior reads retired
        if (g + 1 < nGroups) stage(g + 1);

        const unsigned short* aBuf = aT[g & 1];
        #pragma unroll
        for (int c2 = 0; c2 < 2; ++c2) {
            const int kb = c2 * 32;
            v16bf a0, a1;
            {
                v8bf x0 = *(const v8bf*)&aBuf[m * KGROUP + kb + half * 8];
                v8bf x1 = *(const v8bf*)&aBuf[m * KGROUP + kb + 16 + half * 8];
                v8bf x2 = *(const v8bf*)&aBuf[(16 + m) * KGROUP + kb + half * 8];
                v8bf x3 = *(const v8bf*)&aBuf[(16 + m) * KGROUP + kb + 16 + half * 8];
                #pragma unroll
                for (int i = 0; i < 8; ++i) {
                    a0[i] = x0[i]; a0[i + 8] = x1[i];
                    a1[i] = x2[i]; a1[i + 8] = x3[i];
                }
            }
            const int ch = g * 2 + c2;
            #pragma unroll
            for (int nt = 0; nt < 2; ++nt) {
                const unsigned short* bt =
                    Bpack + ((long)(ch * 256 + ntile0 + nt)) * 512 + lane * 16;
                v16bf b;
                v8bf y0 = *(const v8bf*)(bt);
                v8bf y1 = *(const v8bf*)(bt + 8);
                #pragma unroll
                for (int i = 0; i < 8; ++i) { b[i] = y0[i]; b[i + 8] = y1[i]; }
                if (nt == 0) {
                    acc00 = __builtin_amdgcn_wmma_f32_16x16x32_bf16(
                                false, a0, false, b, (short)0, acc00, false, false);
                    acc10 = __builtin_amdgcn_wmma_f32_16x16x32_bf16(
                                false, a1, false, b, (short)0, acc10, false, false);
                } else {
                    acc01 = __builtin_amdgcn_wmma_f32_16x16x32_bf16(
                                false, a0, false, b, (short)0, acc01, false, false);
                    acc11 = __builtin_amdgcn_wmma_f32_16x16x32_bf16(
                                false, a1, false, b, (short)0, acc11, false, false);
                }
            }
        }
    }

    // C/D layout: VGPR r -> M = r + half*8 ; N = lane%16
    #pragma unroll
    for (int mt = 0; mt < 2; ++mt) {
        const int mrow0 = mbase + mt * 16 + half * 8;
        #pragma unroll
        for (int nt = 0; nt < 2; ++nt) {
            const int n = (ntile0 + nt) * 16 + (lane & 15);
            const float bv = bias[n];
            const v8f acc = mt ? (nt ? acc11 : acc10) : (nt ? acc01 : acc00);
            #pragma unroll
            for (int r = 0; r < 8; ++r)
                gates[(long)(mrow0 + r) * NGATE + n] = acc[r] + bv;
        }
    }
}

// --------------------------------------------------------------------------
// Fused LSTM cell (torch gate order i,f,g,o) + gradient threshold along H.
// Block covers 256 contiguous n of one batch row; neighbor raw values shared
// via LDS; thread 0 recomputes the n-1 neighbor for the block edge.
// c ping-pongs (c_in != c_out); h written fp32 (for FC) and bf16 (for GEMM).
// --------------------------------------------------------------------------
__global__ __launch_bounds__(256)
void lstm_cell_threshold(const float* __restrict__ gates,
                         const float* __restrict__ c_in,
                         const float* __restrict__ mh, const float* __restrict__ vh,
                         const float* __restrict__ mc, const float* __restrict__ vc,
                         float* __restrict__ h_out,
                         unsigned short* __restrict__ hbf_out,
                         float* __restrict__ c_out) {
    __shared__ float shH[257], shC[257];
    const int idx = blockIdx.x * 256 + threadIdx.x;     // BATCH*HDIM
    const int b = idx >> 10, n = idx & 1023;
    const float* g = gates + (long)b * NGATE;

    auto cellraw = [&](int nn, float cin, float& hr, float& cr) {
        float gi = g[nn], gf = g[HDIM + nn], gg = g[2 * HDIM + nn], go = g[3 * HDIM + nn];
        gi = 1.0f / (1.0f + __expf(-gi));
        gf = 1.0f / (1.0f + __expf(-gf));
        go = 1.0f / (1.0f + __expf(-go));
        gg = tanhf(gg);
        cr = gf * cin + gi * gg;
        hr = go * tanhf(cr);
    };

    float hr, cr;
    cellraw(n, c_in[idx], hr, cr);
    shH[threadIdx.x + 1] = hr;
    shC[threadIdx.x + 1] = cr;
    if (threadIdx.x == 0) {
        if (n == 0) { shH[0] = 0.0f; shC[0] = 0.0f; }   // zero-pad at left edge
        else {
            float hp, cp; cellraw(n - 1, c_in[idx - 1], hp, cp);
            shH[0] = hp; shC[0] = cp;
        }
    }
    __syncthreads();

    {
        float gr = hr - shH[threadIdx.x];
        float mm = mh[n], ss = sqrtf(vh[n]);
        float hv = (gr < mm - 3.0f * ss || gr > mm + 3.0f * ss) ? 0.0f : hr;
        h_out[idx] = hv;
        hbf_out[idx] = f2bf(hv);
    }
    {
        float gr = cr - shC[threadIdx.x];
        float mm = mc[n], ss = sqrtf(vc[n]);
        c_out[idx] = (gr < mm - 3.0f * ss || gr > mm + 3.0f * ss) ? 0.0f : cr;
    }
}

// --------------------------------------------------------------------------
// Final FC: out[256,64] = h @ W_fc^T + b_fc  (tiny; plain VALU dot products)
// --------------------------------------------------------------------------
__global__ __launch_bounds__(256)
void fc_out(const float* __restrict__ h, const float* __restrict__ Wfc,
            const float* __restrict__ bfc, float* __restrict__ out) {
    int idx = blockIdx.x * blockDim.x + threadIdx.x;   // BATCH*64
    int b = idx >> 6, o = idx & 63;
    const float4* hp = (const float4*)(h + (long)b * HDIM);
    const float4* wp = (const float4*)(Wfc + (long)o * HDIM);
    float s = 0.0f;
    #pragma unroll 4
    for (int i = 0; i < HDIM / 4; ++i) {
        float4 a = hp[i], w = wp[i];
        s += a.x * w.x + a.y * w.y + a.z * w.z + a.w * w.w;
    }
    out[idx] = s + bfc[o];
}

// --------------------------------------------------------------------------
extern "C" void kernel_launch(void* const* d_in, const int* in_sizes, int n_in,
                              void* d_out, int out_size, void* d_ws, size_t ws_size,
                              hipStream_t stream) {
    const float* x     = (const float*)d_in[0];
    const float* Wih0  = (const float*)d_in[1];
    const float* Whh0  = (const float*)d_in[2];
    const float* bih0  = (const float*)d_in[3];
    const float* bhh0  = (const float*)d_in[4];
    const float* Wih1  = (const float*)d_in[5];
    const float* Whh1  = (const float*)d_in[6];
    const float* bih1  = (const float*)d_in[7];
    const float* bhh1  = (const float*)d_in[8];
    const float* Wfc   = (const float*)d_in[9];
    const float* bfc   = (const float*)d_in[10];
    const float* meanh = (const float*)d_in[11];
    const float* varh  = (const float*)d_in[12];
    const float* meanc = (const float*)d_in[13];
    const float* varc  = (const float*)d_in[14];

    char* ws = (char*)d_ws;
    size_t off = 0;
    auto carve = [&](size_t bytes) {
        char* p = ws + off;
        off = (off + bytes + 255) & ~(size_t)255;
        return p;
    };
    const int K0cat = IDIM + HDIM;                            // 1088
    unsigned short* W0p = (unsigned short*)carve((size_t)K0cat * NGATE * 2);
    unsigned short* W1p = (unsigned short*)carve((size_t)HDIM * NGATE * 2);
    unsigned short* xbf = (unsigned short*)carve((size_t)BATCH * TSTEPS * IDIM * 2);
    unsigned short* hbf = (unsigned short*)carve((size_t)BATCH * HDIM * 2);
    float* bias0 = (float*)carve(NGATE * 4);
    float* bias1 = (float*)carve(NGATE * 4);
    float* gates = (float*)carve((size_t)BATCH * NGATE * 4);
    float* hfp   = (float*)carve((size_t)BATCH * HDIM * 4);
    float* cA    = (float*)carve((size_t)BATCH * HDIM * 4);
    float* cB    = (float*)carve((size_t)BATCH * HDIM * 4);
    (void)ws_size; (void)in_sizes; (void)n_in; (void)out_size;

    // --- per-launch prep ---
    long tot0 = (long)(K0cat >> 5) * 256 * 512;
    long tot1 = (long)(HDIM  >> 5) * 256 * 512;
    pack_w_concat<<<(int)((tot0 + 255) / 256), 256, 0, stream>>>(Wih0, IDIM, Whh0, HDIM, W0p, tot0);
    pack_w_sum   <<<(int)((tot1 + 255) / 256), 256, 0, stream>>>(Wih1, Whh1, HDIM, W1p, tot1);
    {
        int nx = BATCH * TSTEPS * IDIM;
        pack_f32_to_bf16<<<(nx + 255) / 256, 256, 0, stream>>>(x, xbf, nx);
    }
    add_bias_vec<<<NGATE / 256, 256, 0, stream>>>(bih0, bhh0, bias0, NGATE);
    add_bias_vec<<<NGATE / 256, 256, 0, stream>>>(bih1, bhh1, bias1, NGATE);
    zero_u32<<<(BATCH * HDIM / 2) / 256, 256, 0, stream>>>((unsigned int*)hbf, BATCH * HDIM / 2);
    zero_u32<<<(BATCH * HDIM) / 256, 256, 0, stream>>>((unsigned int*)cA, BATCH * HDIM);

    dim3 gGrid(NGATE / 256, BATCH / 32);                      // 16 x 8 blocks
    dim3 gBlk(256);
    const int ewBlocks = (BATCH * HDIM) / 256;                // 1024

    float* cin = cA;
    float* cout = cB;
    for (int t = 0; t < TSTEPS; ++t) {
        // layer 0: gates = [x_t | h](bf16) @ [Wih0|Whh0]^T + (bih0+bhh0)
        gemm_gates<<<gGrid, gBlk, 0, stream>>>(xbf + (long)t * IDIM, TSTEPS * IDIM, IDIM,
                                               hbf, HDIM, HDIM, W0p, bias0, gates);
        lstm_cell_threshold<<<ewBlocks, 256, 0, stream>>>(
            gates, cin, meanh + t * HDIM, varh + t * HDIM,
            meanc + t * HDIM, varc + t * HDIM, hfp, hbf, cout);
        { float* tmp = cin; cin = cout; cout = tmp; }
        // layer 1: x_t == h  =>  gates = h @ (Wih1+Whh1)^T + (bih1+bhh1)
        gemm_gates<<<gGrid, gBlk, 0, stream>>>(hbf, HDIM, HDIM,
                                               hbf, HDIM, 0, W1p, bias1, gates);
        lstm_cell_threshold<<<ewBlocks, 256, 0, stream>>>(
            gates, cin, meanh + t * HDIM, varh + t * HDIM,
            meanc + t * HDIM, varc + t * HDIM, hfp, hbf, cout);
        { float* tmp = cin; cin = cout; cout = tmp; }
    }

    fc_out<<<(BATCH * 64) / 256, 256, 0, stream>>>(hfp, Wfc, bfc, (float*)d_out);
}
